// RAFT_63565515981070
// MI455X (gfx1250) — matrix operand order repaired and checked by reading, not compiled, not run on hardware
//
#include <hip/hip_runtime.h>
#include <math.h>

typedef __attribute__((ext_vector_type(16))) _Float16 v16h;
typedef __attribute__((ext_vector_type(8)))  float    v8f;

#define ELEM_LOOP(idx, total) \
  for (size_t idx = (size_t)blockIdx.x * blockDim.x + threadIdx.x; idx < (total); \
       idx += (size_t)gridDim.x * blockDim.x)

// ---------------------------------------------------------------------------
// WMMA implicit-GEMM:  C[M,N] = act( alpha * A[M,Kp] * B[Kp,N] + bias )
// A: f16 row-major (lda=Kp), B: f16 row-major (ldb=N), C: f32.
// If HWo>0, store in NCHW conv layout: n -> (batch, hw), C[((b*M)+m)*HWo+hw].
// Block = 256 threads = 8 waves; block tile 64(M) x 128(N); wave tile 32x32.
// LDS tiles are stored in WMMA-fragment order so fragment loads are 2x b128.
// ---------------------------------------------------------------------------
__global__ __launch_bounds__(256)
void k_wmma_gemm(const _Float16* __restrict__ A, const _Float16* __restrict__ B,
                 float* __restrict__ C, int M, int N, int Kp, int lda, int ldb,
                 const float* __restrict__ bias, float alpha, int act, int HWo)
{
  __shared__ alignas(32) _Float16 As[4 * 32 * 16];  // 4 subtiles * 32 lanes * 16 halves
  __shared__ alignas(32) _Float16 Bs[8 * 32 * 16];  // 8 subtiles * 32 lanes * 16 halves

  const int t    = threadIdx.x;
  const int lane = t & 31;
  const int wid  = t >> 5;
  const int wm   = wid >> 2;   // 0..1
  const int wn   = wid & 3;    // 0..3
  const int bm   = blockIdx.y * 64;
  const int bn   = blockIdx.x * 128;

  v8f acc[2][2] = {};

  for (int k0 = 0; k0 < Kp; k0 += 32) {
    // ---- stage A (64x32) into fragment order: threads 0..127, one (s,l) each
    if (t < 128) {
      const int s = t >> 5, l = t & 31;
      const int m = bm + s * 16 + (l & 15);
      const int g = l >> 4;
      _Float16* dst = &As[(s * 32 + l) * 16];
      if (m < M) {
        const _Float16* src = &A[(size_t)m * lda + k0];
        if (k0 + 32 < Kp) __builtin_prefetch(src + 32, 0, 1);
#pragma unroll
        for (int h = 0; h < 8; ++h) dst[h] = src[8 * g + h];           // K = 8g+h
#pragma unroll
        for (int h = 0; h < 8; ++h) dst[8 + h] = src[16 + 8 * g + h];  // K = 16+8g+h
      } else {
#pragma unroll
        for (int h = 0; h < 16; ++h) dst[h] = (_Float16)0.0f;
      }
    }
    // ---- stage B (32x128) into fragment order: all 256 threads, one (nt,l) each
    {
      const int nt = t >> 5, l = t & 31;
      const int n  = bn + nt * 16 + (l & 15);
      const int g  = l >> 4;
      _Float16* dst = &Bs[(nt * 32 + l) * 16];
      if (n < N) {
        const _Float16* src = &B[(size_t)(k0 + 16 * g) * ldb + n];
        if (k0 + 32 < Kp) __builtin_prefetch(&B[(size_t)(k0 + 32) * ldb + n], 0, 1);
#pragma unroll
        for (int h = 0; h < 16; ++h) dst[h] = src[(ptrdiff_t)((h < 8 ? h : 8 + h) - 8 * g) * ldb];    // K = 8g+h (h<8), 16+8g+(h-8) (h>=8)
      } else {
#pragma unroll
        for (int h = 0; h < 16; ++h) dst[h] = (_Float16)0.0f;
      }
    }
    __syncthreads();

    const v16h a0 = *(const v16h*)&As[((wm * 2 + 0) * 32 + lane) * 16];
    const v16h a1 = *(const v16h*)&As[((wm * 2 + 1) * 32 + lane) * 16];
    const v16h b0 = *(const v16h*)&Bs[((wn * 2 + 0) * 32 + lane) * 16];
    const v16h b1 = *(const v16h*)&Bs[((wn * 2 + 1) * 32 + lane) * 16];

    acc[0][0] = __builtin_amdgcn_wmma_f32_16x16x32_f16(false, a0, false, b0, (short)0, acc[0][0], false, false);
    acc[0][1] = __builtin_amdgcn_wmma_f32_16x16x32_f16(false, a0, false, b1, (short)0, acc[0][1], false, false);
    acc[1][0] = __builtin_amdgcn_wmma_f32_16x16x32_f16(false, a1, false, b0, (short)0, acc[1][0], false, false);
    acc[1][1] = __builtin_amdgcn_wmma_f32_16x16x32_f16(false, a1, false, b1, (short)0, acc[1][1], false, false);
    __syncthreads();
  }

  // ---- epilogue: C layout per ISA (VGPR r: M=r+8*(lane>=16), N=lane&15)
  for (int i = 0; i < 2; ++i)
    for (int j = 0; j < 2; ++j) {
      const int nIdx = bn + wn * 32 + j * 16 + (lane & 15);
      if (nIdx >= N) continue;
      const int mBase = bm + wm * 32 + i * 16 + 8 * (lane >> 4);
#pragma unroll
      for (int r = 0; r < 8; ++r) {
        const int m = mBase + r;
        if (m >= M) continue;
        float v = acc[i][j][r] * alpha;
        if (bias) v += bias[m];
        if (act == 1) v = fmaxf(v, 0.0f);
        if (HWo > 0) {
          const int b = nIdx / HWo, hw = nIdx % HWo;
          C[((size_t)b * M + m) * (size_t)HWo + hw] = v;
        } else {
          C[(size_t)m * N + nIdx] = v;
        }
      }
    }
}

// ---------------------------------------------------------------------------
// im2col (f32 NCHW -> f16 [Kp][N] row-major, K-remainder zero padded)
// ---------------------------------------------------------------------------
__global__ void k_im2col(const float* __restrict__ x, _Float16* __restrict__ out,
                         int Bn, int Cin, int H, int W, int KH, int stride, int pad,
                         int Ho, int Wo, int Kp)
{
  const size_t N = (size_t)Bn * Ho * Wo;
  const size_t total = (size_t)Kp * N;
  const int K = Cin * KH * KH;
  ELEM_LOOP(idx, total) {
    const int k = (int)(idx / N);
    const size_t n = idx % N;
    float v = 0.0f;
    if (k < K) {
      const int ci = k / (KH * KH);
      const int rem = k % (KH * KH);
      const int rr = rem / KH, cc = rem % KH;
      const int b = (int)(n / (size_t)(Ho * Wo));
      const int hw = (int)(n % (size_t)(Ho * Wo));
      const int oy = hw / Wo, ox = hw % Wo;
      const int iy = oy * stride - pad + rr;
      const int ix = ox * stride - pad + cc;
      if (iy >= 0 && iy < H && ix >= 0 && ix < W)
        v = x[(((size_t)b * Cin + ci) * H + iy) * W + ix];
    }
    out[idx] = (_Float16)v;
  }
}

// weights f32 [M][K] -> f16 [M][Kp], zero padded
__global__ void k_wcvt(const float* __restrict__ w, _Float16* __restrict__ out,
                       int M, int K, int Kp)
{
  const size_t total = (size_t)M * Kp;
  ELEM_LOOP(idx, total) {
    const int m = (int)(idx / Kp), k = (int)(idx % Kp);
    out[idx] = (_Float16)((k < K) ? w[(size_t)m * K + k] : 0.0f);
  }
}

__global__ void k_cvt(const float* __restrict__ in, _Float16* __restrict__ out, size_t n)
{
  ELEM_LOOP(idx, n) out[idx] = (_Float16)in[idx];
}

// dst[m][d] = src[d][HW=m-dim]  (transpose + f16 convert, for corr A = fmap1^T)
__global__ void k_trcvt(const float* __restrict__ src, _Float16* __restrict__ dst,
                        int D, int HW)
{
  const size_t total = (size_t)D * HW;
  ELEM_LOOP(idx, total) {
    const int m = (int)(idx / D), d = (int)(idx % D);
    dst[idx] = (_Float16)src[(size_t)d * HW + m];
  }
}

// ---------------------------------------------------------------------------
// instance norm (biased var), optional fused relu; one block per (b,c) plane
// ---------------------------------------------------------------------------
__global__ __launch_bounds__(256)
void k_instnorm(float* __restrict__ x, int HW, int relu)
{
  float* p = x + (size_t)blockIdx.x * HW;
  __shared__ float s1[256], s2[256];
  float a = 0.0f, b = 0.0f;
  for (int i = threadIdx.x; i < HW; i += 256) { float v = p[i]; a += v; b += v * v; }
  s1[threadIdx.x] = a; s2[threadIdx.x] = b;
  __syncthreads();
  for (int off = 128; off > 0; off >>= 1) {
    if ((int)threadIdx.x < off) { s1[threadIdx.x] += s1[threadIdx.x + off]; s2[threadIdx.x] += s2[threadIdx.x + off]; }
    __syncthreads();
  }
  const float mu = s1[0] / HW;
  const float var = s2[0] / HW - mu * mu;
  const float inv = rsqrtf(var + 1e-5f);
  for (int i = threadIdx.x; i < HW; i += 256) {
    float v = (p[i] - mu) * inv;
    if (relu) v = fmaxf(v, 0.0f);
    p[i] = v;
  }
}

__global__ void k_add_relu(float* __restrict__ out, const float* __restrict__ a,
                           const float* __restrict__ b, size_t n)
{
  ELEM_LOOP(idx, n) out[idx] = fmaxf(a[idx] + b[idx], 0.0f);
}

// 2x2 avg pool over [Q][H][W]
__global__ void k_avgpool(const float* __restrict__ in, float* __restrict__ out,
                          int Q, int H, int W)
{
  const int Ho = H / 2, Wo = W / 2;
  const size_t total = (size_t)Q * Ho * Wo;
  ELEM_LOOP(idx, total) {
    const int wo = (int)(idx % Wo);
    size_t tmp = idx / Wo;
    const int ho = (int)(tmp % Ho);
    const size_t q = tmp / Ho;
    const float* p = in + ((q * H + ho * 2) * (size_t)W) + wo * 2;
    out[idx] = 0.25f * (p[0] + p[1] + p[W] + p[W + 1]);
  }
}

// channel-slice concat: dst[B][Ctot][HW] <- src[B][Cs][HW] at channel offset
__global__ void k_concat(float* __restrict__ dst, int Ctot, int coff,
                         const float* __restrict__ src, int Cs, int HW, int Bn)
{
  const size_t total = (size_t)Bn * Cs * HW;
  ELEM_LOOP(idx, total) {
    const int hw = (int)(idx % HW);
    size_t t = idx / HW;
    const int c = (int)(t % Cs);
    const int b = (int)(t / Cs);
    dst[(((size_t)b * Ctot) + coff + c) * HW + hw] = src[idx];
  }
}

// cnet [B,160,HW] -> net = tanh(ch 0..95), inp = relu(ch 96..159)
__global__ void k_split(const float* __restrict__ cnet, float* __restrict__ net,
                        float* __restrict__ inp, int Bn, int HW)
{
  const size_t total = (size_t)Bn * 160 * HW;
  ELEM_LOOP(idx, total) {
    const int hw = (int)(idx % HW);
    size_t t = idx / HW;
    const int c = (int)(t % 160);
    const int b = (int)(t / 160);
    const float v = cnet[idx];
    if (c < 96) net[((size_t)b * 96 + c) * HW + hw] = tanhf(v);
    else        inp[((size_t)b * 64 + (c - 96)) * HW + hw] = fmaxf(v, 0.0f);
  }
}

__global__ void k_initcoords(float* __restrict__ coords, int Bn, int H, int W)
{
  const int HW = H * W;
  const size_t total = (size_t)Bn * 2 * HW;
  ELEM_LOOP(idx, total) {
    const int hw = (int)(idx % HW);
    const int ch = (int)((idx / HW) % 2);
    coords[idx] = (ch == 0) ? (float)(hw % W) : (float)(hw / W);
  }
}

__global__ void k_flow(const float* __restrict__ coords, float* __restrict__ flow,
                       int Bn, int H, int W)
{
  const int HW = H * W;
  const size_t total = (size_t)Bn * 2 * HW;
  ELEM_LOOP(idx, total) {
    const int hw = (int)(idx % HW);
    const int ch = (int)((idx / HW) % 2);
    const float g = (ch == 0) ? (float)(hw % W) : (float)(hw / W);
    flow[idx] = coords[idx] - g;
  }
}

__global__ void k_addip(float* __restrict__ a, const float* __restrict__ b, size_t n)
{
  ELEM_LOOP(idx, n) a[idx] += b[idx];
}

__device__ __forceinline__ float sigm(float v) { return 1.0f / (1.0f + expf(-v)); }

// hx channels[0..95] = sigmoid(rlin) * net
__global__ void k_rn(float* __restrict__ hx, const float* __restrict__ rlin,
                     const float* __restrict__ net, int Bn, int HW, int Ctot)
{
  const size_t total = (size_t)Bn * 96 * HW;
  ELEM_LOOP(idx, total) {
    const int hw = (int)(idx % HW);
    size_t t = idx / HW;
    const int c = (int)(t % 96);
    const int b = (int)(t / 96);
    hx[(((size_t)b * Ctot) + c) * HW + hw] = sigm(rlin[idx]) * net[idx];
  }
}

// net = (1 - sigmoid(z)) * net + sigmoid(z) * tanh(q)
__global__ void k_gru(float* __restrict__ net, const float* __restrict__ zlin,
                      const float* __restrict__ qlin, size_t n)
{
  ELEM_LOOP(idx, n) {
    const float z = sigm(zlin[idx]);
    net[idx] = (1.0f - z) * net[idx] + z * tanhf(qlin[idx]);
  }
}

// correlation pyramid lookup: out [B, 4*49, HW]; query q = b*HW + hw selects
// the per-pixel [H_l,W_l] slice of level l; bilinear sample with zero padding.
__global__ void k_lookup(const float* __restrict__ c0, const float* __restrict__ c1,
                         const float* __restrict__ c2, const float* __restrict__ c3,
                         const float* __restrict__ coords, float* __restrict__ out,
                         int Bn, int H, int W, int r)
{
  const int HW = H * W;
  const int P = 2 * r + 1, PP = P * P, Cp = 4 * PP;
  const size_t total = (size_t)Bn * Cp * HW;
  ELEM_LOOP(idx, total) {
    const int hw = (int)(idx % HW);
    size_t t = idx / HW;
    const int c = (int)(t % Cp);
    const int b = (int)(t / Cp);
    const int lvl = c / PP;
    const int tt = c % PP;
    const float dx = (float)(tt / P - r);   // added to x (reference semantics)
    const float dy = (float)(tt % P - r);   // added to y
    const float cx = coords[(((size_t)b * 2) + 0) * HW + hw];
    const float cy = coords[(((size_t)b * 2) + 1) * HW + hw];
    const float inv = 1.0f / (float)(1 << lvl);
    const float sx = cx * inv + dx;
    const float sy = cy * inv + dy;
    const int h = H >> lvl, w = W >> lvl;
    const float* base = (lvl == 0) ? c0 : (lvl == 1) ? c1 : (lvl == 2) ? c2 : c3;
    const float* im = base + ((size_t)b * HW + hw) * (size_t)h * w;
    const float x0f = floorf(sx), y0f = floorf(sy);
    const int x0 = (int)x0f, y0 = (int)y0f;
    const float wx = sx - x0f, wy = sy - y0f;
    float acc = 0.0f;
#pragma unroll
    for (int jy = 0; jy < 2; ++jy)
#pragma unroll
      for (int jx = 0; jx < 2; ++jx) {
        const int xi = x0 + jx, yi = y0 + jy;
        const float ww = (jx ? wx : 1.0f - wx) * (jy ? wy : 1.0f - wy);
        if (xi >= 0 && xi < w && yi >= 0 && yi < h) acc += ww * im[(size_t)yi * w + xi];
      }
    out[idx] = acc;
  }
}

// align-corners bilinear resize [B,2,H,W] -> [B,2,Ho,Wo], times scale
__global__ void k_resize(const float* __restrict__ flow, float* __restrict__ out,
                         int Bn, int H, int W, int Ho, int Wo, float scale)
{
  const size_t total = (size_t)Bn * 2 * Ho * Wo;
  ELEM_LOOP(idx, total) {
    const int x = (int)(idx % Wo);
    size_t t = idx / Wo;
    const int y = (int)(t % Ho);
    t /= Ho;
    const int c = (int)(t % 2);
    const int b = (int)(t / 2);
    const float fy = (Ho > 1) ? (float)y * (H - 1.0f) / (Ho - 1.0f) : 0.0f;
    const float fx = (Wo > 1) ? (float)x * (W - 1.0f) / (Wo - 1.0f) : 0.0f;
    const int y0 = (int)floorf(fy); const int y1 = min(y0 + 1, H - 1);
    const int x0 = (int)floorf(fx); const int x1 = min(x0 + 1, W - 1);
    const float wy = fy - y0, wx = fx - x0;
    const float* p = flow + (((size_t)b * 2 + c) * H) * (size_t)W;
    const float v = p[(size_t)y0 * W + x0] * (1 - wy) * (1 - wx)
                  + p[(size_t)y1 * W + x0] * wy * (1 - wx)
                  + p[(size_t)y0 * W + x1] * (1 - wy) * wx
                  + p[(size_t)y1 * W + x1] * wy * wx;
    out[idx] = scale * v;
  }
}

// ---------------------------------------------------------------------------
// host side
// ---------------------------------------------------------------------------
struct CW { const float* w; const float* bias; int co, ci, k, K, Kp; _Float16* w16; };
struct EncP { CW conv1; CW b[6][4]; bool ds[6]; CW conv2; };

static inline unsigned gridFor(size_t total)
{
  size_t b = (total + 255) / 256;
  if (b < 1) b = 1;
  if (b > 1048576) b = 1048576;
  return (unsigned)b;
}

extern "C" void kernel_launch(void* const* d_in, const int* in_sizes, int n_in,
                              void* d_out, int out_size, void* d_ws, size_t ws_size,
                              hipStream_t stream)
{
  (void)in_sizes; (void)n_in; (void)out_size; (void)ws_size;
  const size_t F = sizeof(float);

  // ---- workspace bump allocator
  size_t cur = 0;
  auto alloc = [&](size_t bytes) -> void* {
    cur = (cur + 255) & ~(size_t)255;
    void* p = (char*)d_ws + cur;
    cur += bytes;
    return p;
  };

  // ---- parameter walk (setup_inputs insertion order)
  int di = 2;
  auto nextConv = [&](int co, int ci, int k) -> CW {
    CW c; c.w = (const float*)d_in[di++]; c.bias = (const float*)d_in[di++];
    c.co = co; c.ci = ci; c.k = k; c.K = ci * k * k; c.Kp = (c.K + 31) & ~31;
    c.w16 = nullptr; return c;
  };
  const int cins[6] = {32, 32, 32, 64, 64, 96};
  const int pls[6]  = {32, 32, 64, 64, 96, 96};
  const int ss[6]   = {1, 1, 2, 1, 2, 1};
  auto loadEnc = [&](int outdim) -> EncP {
    EncP e;
    e.conv1 = nextConv(32, 3, 7);
    for (int i = 0; i < 6; ++i) {
      e.b[i][0] = nextConv(pls[i] / 4, cins[i], 1);
      e.b[i][1] = nextConv(pls[i] / 4, pls[i] / 4, 3);
      e.b[i][2] = nextConv(pls[i], pls[i] / 4, 1);
      e.ds[i] = (ss[i] != 1);
      if (e.ds[i]) e.b[i][3] = nextConv(pls[i], cins[i], 1);
    }
    e.conv2 = nextConv(outdim, 96, 1);
    return e;
  };
  EncP fnet = loadEnc(128);
  EncP cnet = loadEnc(160);
  CW convc1 = nextConv(96, 196, 1);
  CW convf1 = nextConv(64, 2, 7);
  CW convf2 = nextConv(32, 64, 3);
  CW convm  = nextConv(80, 128, 3);
  CW convz  = nextConv(96, 242, 3);
  CW convr  = nextConv(96, 242, 3);
  CW convq  = nextConv(96, 242, 3);
  CW fh1    = nextConv(128, 96, 3);
  CW fh2    = nextConv(2, 128, 3);
  // d_in[di] == iters (hardcoded to 12 below)

  // ---- buffers
  const int H8 = 48, W8 = 96, HW8 = H8 * W8;         // 4608
  float* imgcat = (float*)alloc((size_t)4 * 3 * 384 * 768 * F);
  float* bufA   = (float*)alloc((size_t)9437184 * F);
  float* bufB   = (float*)alloc((size_t)9437184 * F);
  float* ybuf   = (float*)alloc((size_t)9437184 * F);
  float* t1     = (float*)alloc((size_t)4718592 * F);
  float* t2     = (float*)alloc((size_t)4718592 * F);
  float* dsbuf  = (float*)alloc((size_t)4718592 * F);
  _Float16* imb = (_Float16*)alloc((size_t)47185920 * 2);
  float* fmapb  = (float*)alloc((size_t)4 * 128 * HW8 * F);
  float* cnetb  = (float*)alloc((size_t)2 * 160 * HW8 * F);
  _Float16* A16 = (_Float16*)alloc((size_t)HW8 * 128 * 2);
  _Float16* B16 = (_Float16*)alloc((size_t)HW8 * 128 * 2);
  float* corr0  = (float*)alloc((size_t)2 * HW8 * HW8 * F);
  float* corr1  = (float*)alloc((size_t)2 * HW8 * 1152 * F);
  float* corr2  = (float*)alloc((size_t)2 * HW8 * 288 * F);
  float* corr3  = (float*)alloc((size_t)2 * HW8 * 72 * F);
  float* net    = (float*)alloc((size_t)2 * 96 * HW8 * F);
  float* inp    = (float*)alloc((size_t)2 * 64 * HW8 * F);
  float* coords = (float*)alloc((size_t)2 * 2 * HW8 * F);
  float* flowb  = (float*)alloc((size_t)2 * 2 * HW8 * F);
  float* corrf  = (float*)alloc((size_t)2 * 196 * HW8 * F);
  float* corb   = (float*)alloc((size_t)2 * 96 * HW8 * F);
  float* flo1   = (float*)alloc((size_t)2 * 64 * HW8 * F);
  float* flo2   = (float*)alloc((size_t)2 * 32 * HW8 * F);
  float* cat128 = (float*)alloc((size_t)2 * 128 * HW8 * F);
  float* motb   = (float*)alloc((size_t)2 * 80 * HW8 * F);
  float* hx     = (float*)alloc((size_t)2 * 242 * HW8 * F);
  float* zlin   = (float*)alloc((size_t)2 * 96 * HW8 * F);
  float* rlin   = (float*)alloc((size_t)2 * 96 * HW8 * F);
  float* qlin   = (float*)alloc((size_t)2 * 96 * HW8 * F);
  float* fhb    = (float*)alloc((size_t)2 * 128 * HW8 * F);
  float* dfb    = (float*)alloc((size_t)2 * 2 * HW8 * F);

  // ---- convert all weights to K-padded f16 once
  CW* all[64]; int na = 0;
  auto regEnc = [&](EncP& e) {
    all[na++] = &e.conv1;
    for (int i = 0; i < 6; ++i) {
      all[na++] = &e.b[i][0]; all[na++] = &e.b[i][1]; all[na++] = &e.b[i][2];
      if (e.ds[i]) all[na++] = &e.b[i][3];
    }
    all[na++] = &e.conv2;
  };
  regEnc(fnet); regEnc(cnet);
  all[na++] = &convc1; all[na++] = &convf1; all[na++] = &convf2; all[na++] = &convm;
  all[na++] = &convz;  all[na++] = &convr;  all[na++] = &convq;
  all[na++] = &fh1;    all[na++] = &fh2;
  for (int i = 0; i < na; ++i) {
    CW* c = all[i];
    c->w16 = (_Float16*)alloc((size_t)c->co * c->Kp * 2);
    k_wcvt<<<gridFor((size_t)c->co * c->Kp), 256, 0, stream>>>(c->w, c->w16, c->co, c->K, c->Kp);
  }

  // ---- generic conv: im2col(f16) + WMMA GEMM, fused bias/act, NCHW store
  auto runConv = [&](const float* x, float* y, const CW& c, int Bn, int H, int W,
                     int stride, int pad, int act, int* pHo = nullptr, int* pWo = nullptr) {
    const int Ho = (H + 2 * pad - c.k) / stride + 1;
    const int Wo = (W + 2 * pad - c.k) / stride + 1;
    const size_t N = (size_t)Bn * Ho * Wo;
    k_im2col<<<gridFor((size_t)c.Kp * N), 256, 0, stream>>>(
        x, imb, Bn, c.ci, H, W, c.k, stride, pad, Ho, Wo, c.Kp);
    dim3 g((unsigned)((N + 127) / 128), (unsigned)((c.co + 63) / 64));
    k_wmma_gemm<<<g, 256, 0, stream>>>(c.w16, imb, y, c.co, (int)N, c.Kp, c.Kp, (int)N,
                                       c.bias, 1.0f, act, Ho * Wo);
    if (pHo) *pHo = Ho;
    if (pWo) *pWo = Wo;
  };

  // ---- encoder (fnet: instance norm; cnet: no norm)
  auto runEnc = [&](const float* img, int Bn, EncP& e, bool norm, float* outbuf) {
    int H, W;
    runConv(img, bufA, e.conv1, Bn, 384, 768, 2, 3, norm ? 0 : 1, &H, &W);
    if (norm) k_instnorm<<<Bn * 32, 256, 0, stream>>>(bufA, H * W, 1);
    float* x = bufA;
    for (int i = 0; i < 6; ++i) {
      const int pl = pls[i], q = pl / 4, s = ss[i];
      int H2, W2;
      runConv(x, t1, e.b[i][0], Bn, H, W, 1, 0, norm ? 0 : 1);
      if (norm) k_instnorm<<<Bn * q, 256, 0, stream>>>(t1, H * W, 1);
      runConv(t1, t2, e.b[i][1], Bn, H, W, s, 1, norm ? 0 : 1, &H2, &W2);
      if (norm) k_instnorm<<<Bn * q, 256, 0, stream>>>(t2, H2 * W2, 1);
      runConv(t2, ybuf, e.b[i][2], Bn, H2, W2, 1, 0, norm ? 0 : 1);
      if (norm) k_instnorm<<<Bn * pl, 256, 0, stream>>>(ybuf, H2 * W2, 1);
      const float* xd = x;
      if (e.ds[i]) {
        runConv(x, dsbuf, e.b[i][3], Bn, H, W, s, 0, 0);
        if (norm) k_instnorm<<<Bn * pl, 256, 0, stream>>>(dsbuf, H2 * W2, 0);
        xd = dsbuf;
      }
      float* xn = (x == bufA) ? bufB : bufA;
      const size_t n = (size_t)Bn * pl * H2 * W2;
      k_add_relu<<<gridFor(n), 256, 0, stream>>>(xn, xd, ybuf, n);
      x = xn; H = H2; W = W2;
    }
    runConv(x, outbuf, e.conv2, Bn, H, W, 1, 0, 0);
  };

  // ---- fnet on concat(image1, image2) along batch; cnet on image1
  const size_t imgElems = (size_t)2 * 3 * 384 * 768;
  hipMemcpyAsync(imgcat, d_in[0], imgElems * F, hipMemcpyDeviceToDevice, stream);
  hipMemcpyAsync(imgcat + imgElems, d_in[1], imgElems * F, hipMemcpyDeviceToDevice, stream);
  runEnc(imgcat, 4, fnet, true, fmapb);
  runEnc((const float*)d_in[0], 2, cnet, false, cnetb);

  // ---- correlation pyramid: per batch, corr = (fmap1^T · fmap2) / sqrt(128)
  const float rs = 1.0f / sqrtf(128.0f);
  for (int b = 0; b < 2; ++b) {
    k_trcvt<<<gridFor((size_t)HW8 * 128), 256, 0, stream>>>(
        fmapb + (size_t)b * 128 * HW8, A16, 128, HW8);
    k_cvt<<<gridFor((size_t)128 * HW8), 256, 0, stream>>>(
        fmapb + (size_t)(2 + b) * 128 * HW8, B16, (size_t)128 * HW8);
    dim3 g((unsigned)((HW8 + 127) / 128), (unsigned)((HW8 + 63) / 64));
    k_wmma_gemm<<<g, 256, 0, stream>>>(A16, B16, corr0 + (size_t)b * HW8 * HW8,
                                       HW8, HW8, 128, 128, HW8, nullptr, rs, 0, 0);
  }
  const int Q = 2 * HW8;
  k_avgpool<<<gridFor((size_t)Q * 24 * 48), 256, 0, stream>>>(corr0, corr1, Q, 48, 96);
  k_avgpool<<<gridFor((size_t)Q * 12 * 24), 256, 0, stream>>>(corr1, corr2, Q, 24, 48);
  k_avgpool<<<gridFor((size_t)Q * 6 * 12), 256, 0, stream>>>(corr2, corr3, Q, 12, 24);

  // ---- context split + coords init
  k_split<<<gridFor((size_t)2 * 160 * HW8), 256, 0, stream>>>(cnetb, net, inp, 2, HW8);
  k_initcoords<<<gridFor((size_t)2 * 2 * HW8), 256, 0, stream>>>(coords, 2, H8, W8);

  // ---- 12 GRU update iterations
  const size_t n96 = (size_t)2 * 96 * HW8;
  for (int it = 0; it < 12; ++it) {
    k_lookup<<<gridFor((size_t)2 * 196 * HW8), 256, 0, stream>>>(
        corr0, corr1, corr2, corr3, coords, corrf, 2, H8, W8, 3);
    k_flow<<<gridFor((size_t)2 * 2 * HW8), 256, 0, stream>>>(coords, flowb, 2, H8, W8);

    runConv(corrf, corb, convc1, 2, H8, W8, 1, 0, 1);
    runConv(flowb, flo1, convf1, 2, H8, W8, 1, 3, 1);
    runConv(flo1, flo2, convf2, 2, H8, W8, 1, 1, 1);
    k_concat<<<gridFor(n96), 256, 0, stream>>>(cat128, 128, 0, corb, 96, HW8, 2);
    k_concat<<<gridFor((size_t)2 * 32 * HW8), 256, 0, stream>>>(cat128, 128, 96, flo2, 32, HW8, 2);
    runConv(cat128, motb, convm, 2, H8, W8, 1, 1, 1);

    // hx = [net 96 | inp 64 | mot 80 | flow 2] = 242 channels
    k_concat<<<gridFor(n96), 256, 0, stream>>>(hx, 242, 0, net, 96, HW8, 2);
    k_concat<<<gridFor((size_t)2 * 64 * HW8), 256, 0, stream>>>(hx, 242, 96, inp, 64, HW8, 2);
    k_concat<<<gridFor((size_t)2 * 80 * HW8), 256, 0, stream>>>(hx, 242, 160, motb, 80, HW8, 2);
    k_concat<<<gridFor((size_t)2 * 2 * HW8), 256, 0, stream>>>(hx, 242, 240, flowb, 2, HW8, 2);

    // z and r share one im2col of hx
    {
      const size_t N = (size_t)2 * HW8;
      k_im2col<<<gridFor((size_t)convz.Kp * N), 256, 0, stream>>>(
          hx, imb, 2, 242, H8, W8, 3, 1, 1, H8, W8, convz.Kp);
      dim3 g((unsigned)((N + 127) / 128), (unsigned)((96 + 63) / 64));
      k_wmma_gemm<<<g, 256, 0, stream>>>(convz.w16, imb, zlin, 96, (int)N, convz.Kp,
                                         convz.Kp, (int)N, convz.bias, 1.0f, 0, HW8);
      k_wmma_gemm<<<g, 256, 0, stream>>>(convr.w16, imb, rlin, 96, (int)N, convr.Kp,
                                         convr.Kp, (int)N, convr.bias, 1.0f, 0, HW8);
    }
    // q: replace hx[0..95] with sigmoid(r)*net, re-im2col
    k_rn<<<gridFor(n96), 256, 0, stream>>>(hx, rlin, net, 2, HW8, 242);
    runConv(hx, qlin, convq, 2, H8, W8, 1, 1, 0);
    k_gru<<<gridFor(n96), 256, 0, stream>>>(net, zlin, qlin, n96);

    runConv(net, fhb, fh1, 2, H8, W8, 1, 1, 1);
    runConv(fhb, dfb, fh2, 2, H8, W8, 1, 1, 0);
    k_addip<<<gridFor((size_t)2 * 2 * HW8), 256, 0, stream>>>(coords, dfb, (size_t)2 * 2 * HW8);
  }

  // ---- final flow, upsample x2 (align corners) and scale by 2
  k_flow<<<gridFor((size_t)2 * 2 * HW8), 256, 0, stream>>>(coords, flowb, 2, H8, W8);
  k_resize<<<gridFor((size_t)2 * 2 * 96 * 192), 256, 0, stream>>>(
      flowb, (float*)d_out, 2, H8, W8, 96, 192, 2.0f);
}
